// TransformerDecoderLayer_8993661518319
// MI455X (gfx1250) — matrix operand order repaired and checked by reading, not compiled, not run on hardware
//
#include <hip/hip_runtime.h>

// ---------------------------------------------------------------------------
// Transformer decoder layer for MI455X (gfx1250).
// bf16 WMMA + Tensor Data Mover (TDM) staging, fp32 accumulate.
// N=4, S=T=1024, E=1024, NH=16, HD=64, DFF=4096, fp32 I/O.
// ---------------------------------------------------------------------------

#define E_DIM   1024
#define NH_DIM  16
#define HD_DIM  64
#define DFF_DIM 4096
#define NB_DIM  4
#define S_DIM   1024
#define T_DIM   1024

typedef __attribute__((ext_vector_type(16))) __bf16 v16bf;
typedef __attribute__((ext_vector_type(8)))  float  v8f;
typedef __attribute__((ext_vector_type(4)))  unsigned int v4u;
typedef __attribute__((ext_vector_type(8)))  int    v8i;
typedef __attribute__((ext_vector_type(4)))  int    v4i;
typedef unsigned short u16;
typedef unsigned int   u32;
typedef unsigned long long u64;

static __device__ __forceinline__ u16 f2bf(float f) {
  u32 u = __float_as_uint(f);
  return (u16)((u + 0x7FFFu + ((u >> 16) & 1u)) >> 16);   // RNE
}

static __device__ __forceinline__ v16bf make_frag(uint4 lo, uint4 hi) {
  union { uint4 u[2]; v16bf v; } c;
  c.u[0] = lo; c.u[1] = hi;
  return c.v;
}

static __device__ __forceinline__ v8f wmma_bf16(v16bf a, v16bf b, v8f c) {
  return __builtin_amdgcn_wmma_f32_16x16x32_bf16(
      false, a, false, b, (short)0, c, false, false);
}

// ---------------------------------------------------------------------------
// TDM: load a 128(rows) x 32(cols) bf16 tile, row stride = stride_elems,
// from global to LDS (rows packed at 64B pitch).  D# per CDNA5 ISA 8.3/8.4.
// tensor_dim == tile_dim -> never OOB (tiles are in-bounds by construction).
// Toolchain uses the 6-arg builtin: (g0 v4u, g1 v8i, v4i, v4i, v8i, cpol).
// ---------------------------------------------------------------------------
static __device__ __forceinline__ void tdm_load_tile_128x32(
    u32 lds_off, const u16* gptr, u32 stride_elems) {
  const u64 ga = (u64)(uintptr_t)gptr;
  v4u g0;
  g0.x = 1u;                                            // count=1, user D#
  g0.y = lds_off;                                       // lds_addr
  g0.z = (u32)ga;                                       // global_addr[31:0]
  g0.w = ((u32)(ga >> 32) & 0x01FFFFFFu) | 0x80000000u; // addr[56:32] | type=2
  const u32 d0 = 32u;      // tensor_dim0 (elements)
  const u32 d1 = 128u;     // tensor_dim1 (rows)
  const u64 s0 = (u64)stride_elems;  // tensor_dim0_stride (elements)
  v8i g1;
  g1[0] = (int)(1u << 16);                              // data_size=1 (2B)
  g1[1] = (int)((d0 & 0xFFFFu) << 16);                  // dim0 lo16 @ [63:48]
  g1[2] = (int)((d0 >> 16) | ((d1 & 0xFFFFu) << 16));   // dim0 hi | dim1 lo
  g1[3] = (int)((d1 >> 16) | (32u << 16));              // dim1 hi | tile_dim0
  g1[4] = (int)128u;                                    // tile_dim1 (tile_dim2=0)
  g1[5] = (int)(u32)s0;                                 // stride0[31:0]
  g1[6] = (int)((u32)(s0 >> 32) & 0xFFFFu);             // stride0[47:32]
  g1[7] = 0;
  const v4i z4 = {0, 0, 0, 0};
  const v8i z8 = {0, 0, 0, 0, 0, 0, 0, 0};
  __builtin_amdgcn_tensor_load_to_lds(g0, g1, z4, z4, z8, 0);
}

// ---------------------------------------------------------------------------
// GEMM: C[M,N] = act(A[M,K](bf16) @ WT[N,K](bf16, pre-transposed) + bias).
// Block 128x128, K-step 32, double-buffered TDM staging, 8 waves.
// ---------------------------------------------------------------------------
template <bool RELU, bool OUT_BF16>
__global__ __launch_bounds__(256) void gemm_tdm(
    const u16* __restrict__ A, const u16* __restrict__ WT,
    const float* __restrict__ bias, void* __restrict__ Cout,
    int M, int Nn, int K) {
  __shared__ u16 As[2][128][32];
  __shared__ u16 Ws[2][128][32];

  const int tid  = threadIdx.x;
  const int wave = tid >> 5;
  const int lane = tid & 31;
  const int half = lane >> 4;
  const int l16  = lane & 15;
  const int wr   = wave & 3;
  const int wc   = wave >> 2;
  const int nbase = blockIdx.x * 128;
  const int mbase = blockIdx.y * 128;

  v8f acc[2][4];
  const v8f z = {0.f, 0.f, 0.f, 0.f, 0.f, 0.f, 0.f, 0.f};
#pragma unroll
  for (int i = 0; i < 2; ++i)
#pragma unroll
    for (int j = 0; j < 4; ++j) acc[i][j] = z;

  const u32 ldsA[2] = {(u32)(uintptr_t)&As[0][0][0], (u32)(uintptr_t)&As[1][0][0]};
  const u32 ldsW[2] = {(u32)(uintptr_t)&Ws[0][0][0], (u32)(uintptr_t)&Ws[1][0][0]};
  const u16* Ab = A  + (size_t)mbase * K;
  const u16* Wb = WT + (size_t)nbase * K;

  if (wave == 0) {
    tdm_load_tile_128x32(ldsA[0], Ab, (u32)K);
    tdm_load_tile_128x32(ldsW[0], Wb, (u32)K);
  }

  int buf = 0;
  for (int k0 = 0; k0 < K; k0 += 32) {
    if (wave == 0) __builtin_amdgcn_s_wait_tensorcnt(0);
    __syncthreads();                      // current tiles visible to all waves
    if (wave == 0 && (k0 + 32) < K) {     // prefetch next tiles into other buf
      tdm_load_tile_128x32(ldsA[buf ^ 1], Ab + k0 + 32, (u32)K);
      tdm_load_tile_128x32(ldsW[buf ^ 1], Wb + k0 + 32, (u32)K);
    }
    const u16 (*Ac)[32] = As[buf];
    const u16 (*Wc)[32] = Ws[buf];
    // A frags: 16-bit A layout -> lanes<16: +0/+32B, lanes>=16: +16/+48B
    v16bf af[2];
#pragma unroll
    for (int i = 0; i < 2; ++i) {
      const uint4* p = (const uint4*)&Ac[wr * 32 + i * 16 + l16][half * 8];
      af[i] = make_frag(p[0], p[2]);
    }
    // B frags: same 16-bit layout as A -> lanes<16: +0/+32B, lanes>=16: +16/+48B
#pragma unroll
    for (int j = 0; j < 4; ++j) {
      const uint4* p = (const uint4*)&Wc[wc * 64 + j * 16 + l16][half * 8];
      v16bf bf = make_frag(p[0], p[2]);
#pragma unroll
      for (int i = 0; i < 2; ++i) acc[i][j] = wmma_bf16(af[i], bf, acc[i][j]);
    }
    __syncthreads();                      // done reading buf before reuse
    buf ^= 1;
  }

  // epilogue: C layout VGPR r -> row (i*16 + half*8 + r), col = l16 lane
#pragma unroll
  for (int i = 0; i < 2; ++i) {
    int mrow = mbase + wr * 32 + i * 16 + half * 8;
#pragma unroll
    for (int j = 0; j < 4; ++j) {
      int col = nbase + wc * 64 + j * 16 + l16;
      float bv = bias[col];
#pragma unroll
      for (int r = 0; r < 8; ++r) {
        float v = acc[i][j][r] + bv;
        if (RELU) v = fmaxf(v, 0.f);
        if (OUT_BF16) ((u16*)Cout)[(size_t)(mrow + r) * Nn + col] = f2bf(v);
        else          ((float*)Cout)[(size_t)(mrow + r) * Nn + col] = v;
      }
    }
  }
}

// ---------------------------------------------------------------------------
// Flash attention, bf16 in / bf16 out.  Block = (batch, head, 64 Q rows),
// 4 waves x 16 rows.  KV streamed in 32-col chunks, online softmax.
// ---------------------------------------------------------------------------
template <bool CAUSAL>
__global__ __launch_bounds__(128) void attn_kernel(
    const u16* __restrict__ Q, const u16* __restrict__ Kb,
    const u16* __restrict__ Vb, u16* __restrict__ O, int Sq, int Tk) {
  __shared__ u16 Qs[64][64];
  __shared__ u16 Kl[32][64];
  __shared__ u16 Vt[64][32];
  __shared__ u16 Pl[4][16][32];

  const int n  = blockIdx.z, h = blockIdx.y;
  const int qb = blockIdx.x * 64;
  const int tid  = threadIdx.x;
  const int w    = tid >> 5;
  const int lane = tid & 31;
  const int half = lane >> 4;
  const int l16  = lane & 15;

  const u16* Qp = Q  + ((size_t)n * Sq + qb) * E_DIM + h * HD_DIM;
  const u16* Kp = Kb + (size_t)n * Tk * E_DIM + h * HD_DIM;
  const u16* Vp = Vb + (size_t)n * Tk * E_DIM + h * HD_DIM;

  // stage Q (64x64 bf16): 512 uint4, pure copy
#pragma unroll
  for (int i = 0; i < 4; ++i) {
    int idx = tid + i * 128;
    int row = idx >> 3, cv = (idx & 7) << 3;
    *(uint4*)&Qs[row][cv] = *(const uint4*)(Qp + (size_t)row * E_DIM + cv);
  }
  __syncthreads();

  v16bf qf[2];
#pragma unroll
  for (int c = 0; c < 2; ++c) {
    const uint4* p = (const uint4*)&Qs[w * 16 + l16][c * 32 + half * 8];
    qf[c] = make_frag(p[0], p[2]);
  }

  const v8f z = {0.f, 0.f, 0.f, 0.f, 0.f, 0.f, 0.f, 0.f};
  v8f o[4] = {z, z, z, z};
  float mrow[8], lrow[8];
#pragma unroll
  for (int r = 0; r < 8; ++r) { mrow[r] = -1e30f; lrow[r] = 0.f; }

  const int tlimit = CAUSAL ? (qb + 64) : Tk;
  for (int t0 = 0; t0 < tlimit; t0 += 32) {
    // stage K chunk (straight copy) and V chunk (transposed)
#pragma unroll
    for (int i = 0; i < 2; ++i) {
      int idx = tid + i * 128;
      int row = idx >> 3, cv = (idx & 7) << 3;
      *(uint4*)&Kl[row][cv] = *(const uint4*)(Kp + (size_t)(t0 + row) * E_DIM + cv);
      uint4 fv = *(const uint4*)(Vp + (size_t)(t0 + row) * E_DIM + cv);
      const u16* pv = (const u16*)&fv;
#pragma unroll
      for (int e = 0; e < 8; ++e) Vt[cv + e][row] = pv[e];
    }
    __syncthreads();

    // S = Q K^T (two 16-col tiles, 2 d-chunks)
    v8f s0 = z, s1 = z;
#pragma unroll
    for (int c = 0; c < 2; ++c) {
      const uint4* p0 = (const uint4*)&Kl[l16][c * 32 + half * 8];
      const uint4* p1 = (const uint4*)&Kl[16 + l16][c * 32 + half * 8];
      s0 = wmma_bf16(qf[c], make_frag(p0[0], p0[2]), s0);
      s1 = wmma_bf16(qf[c], make_frag(p1[0], p1[2]), s1);
    }
#pragma unroll
    for (int r = 0; r < 8; ++r) {
      float a0 = s0[r] * 0.125f, a1 = s1[r] * 0.125f;
      if (CAUSAL) {
        int rowg = qb + w * 16 + half * 8 + r;
        if (t0 + l16 > rowg)      a0 = -1e30f;
        if (t0 + 16 + l16 > rowg) a1 = -1e30f;
      }
      s0[r] = a0; s1[r] = a1;
    }
    // online softmax (reductions inside 16-lane halves = C-layout rows)
#pragma unroll
    for (int r = 0; r < 8; ++r) {
      float mx = fmaxf(s0[r], s1[r]);
#pragma unroll
      for (int off = 1; off < 16; off <<= 1)
        mx = fmaxf(mx, __shfl_xor(mx, off, 32));
      float mn = fmaxf(mrow[r], mx);
      float alpha = __expf(mrow[r] - mn);
      mrow[r] = mn;
      float p0 = __expf(s0[r] - mn);
      float p1 = __expf(s1[r] - mn);
      s0[r] = p0; s1[r] = p1;
      float sum = p0 + p1;
#pragma unroll
      for (int off = 1; off < 16; off <<= 1)
        sum += __shfl_xor(sum, off, 32);
      lrow[r] = lrow[r] * alpha + sum;
#pragma unroll
      for (int j = 0; j < 4; ++j) o[j][r] *= alpha;
    }
    // P: C layout -> LDS -> A layout
#pragma unroll
    for (int r = 0; r < 8; ++r) {
      Pl[w][half * 8 + r][l16]      = f2bf(s0[r]);
      Pl[w][half * 8 + r][16 + l16] = f2bf(s1[r]);
    }
    const uint4* pp = (const uint4*)&Pl[w][l16][half * 8];
    v16bf pf = make_frag(pp[0], pp[2]);
#pragma unroll
    for (int j = 0; j < 4; ++j) {
      const uint4* pv = (const uint4*)&Vt[j * 16 + l16][half * 8];
      o[j] = wmma_bf16(pf, make_frag(pv[0], pv[2]), o[j]);
    }
    __syncthreads();
  }

#pragma unroll
  for (int r = 0; r < 8; ++r) {
    float inv = 1.f / lrow[r];
    int rowg = qb + w * 16 + half * 8 + r;
#pragma unroll
    for (int j = 0; j < 4; ++j) {
      int col = h * HD_DIM + j * 16 + l16;
      O[((size_t)n * Sq + rowg) * E_DIM + col] = f2bf(o[j][r] * inv);
    }
  }
}

// ---------------------------------------------------------------------------
// Residual + LayerNorm; optional dual output (fp32 + bf16 for next GEMM).
// ---------------------------------------------------------------------------
template <bool DUAL>
__global__ __launch_bounds__(256) void add_ln(
    const float* __restrict__ a, const float* __restrict__ b,
    const float* __restrict__ g, const float* __restrict__ be,
    float* __restrict__ out, u16* __restrict__ outb) {
  __shared__ float red[16];
  __shared__ float stats[2];
  const int tid = threadIdx.x;
  const size_t base = (size_t)blockIdx.x * E_DIM;
  const int c = tid * 4;

  float4 va = *(const float4*)(a + base + c);
  float4 vb = *(const float4*)(b + base + c);
  float v0 = va.x + vb.x, v1 = va.y + vb.y, v2 = va.z + vb.z, v3 = va.w + vb.w;
  float s  = v0 + v1 + v2 + v3;
  float ss = v0 * v0 + v1 * v1 + v2 * v2 + v3 * v3;
#pragma unroll
  for (int off = 1; off < 32; off <<= 1) {
    s  += __shfl_xor(s, off, 32);
    ss += __shfl_xor(ss, off, 32);
  }
  if ((tid & 31) == 0) { red[tid >> 5] = s; red[8 + (tid >> 5)] = ss; }
  __syncthreads();
  if (tid == 0) {
    float S = 0.f, SS = 0.f;
#pragma unroll
    for (int wv = 0; wv < 8; ++wv) { S += red[wv]; SS += red[8 + wv]; }
    float mu  = S * (1.f / E_DIM);
    float var = SS * (1.f / E_DIM) - mu * mu;
    stats[0] = mu;
    stats[1] = rsqrtf(var + 1e-5f);
  }
  __syncthreads();
  float mu = stats[0], rs = stats[1];
  float4 vg  = *(const float4*)(g + c);
  float4 vbe = *(const float4*)(be + c);
  float4 ot;
  ot.x = (v0 - mu) * rs * vg.x + vbe.x;
  ot.y = (v1 - mu) * rs * vg.y + vbe.y;
  ot.z = (v2 - mu) * rs * vg.z + vbe.z;
  ot.w = (v3 - mu) * rs * vg.w + vbe.w;
  *(float4*)(out + base + c) = ot;
  if (DUAL) {
    u32 p0 = (u32)f2bf(ot.x) | ((u32)f2bf(ot.y) << 16);
    u32 p1 = (u32)f2bf(ot.z) | ((u32)f2bf(ot.w) << 16);
    *(uint2*)(outb + base + c) = make_uint2(p0, p1);
  }
}

// ---------------------------------------------------------------------------
// One-time converters: fp32 -> bf16 (elementwise) and fp32 W[K,N] -> bf16
// WT[N,K] (tiled transpose).
// ---------------------------------------------------------------------------
__global__ __launch_bounds__(256) void convert_bf16(
    const float* __restrict__ in, u16* __restrict__ out) {
  const size_t c = ((size_t)blockIdx.x * 256 + threadIdx.x) * 4;
  float4 f = *(const float4*)(in + c);
  u32 p0 = (u32)f2bf(f.x) | ((u32)f2bf(f.y) << 16);
  u32 p1 = (u32)f2bf(f.z) | ((u32)f2bf(f.w) << 16);
  *(uint2*)(out + c) = make_uint2(p0, p1);
}

__global__ __launch_bounds__(256) void transpose_convert(
    const float* __restrict__ W, u16* __restrict__ WT, int K, int Nn) {
  __shared__ float tile[32][33];
  const int kb = blockIdx.x * 32, nb = blockIdx.y * 32;
  const int r  = threadIdx.x >> 3;
  const int c4 = (threadIdx.x & 7) << 2;
  float4 f = *(const float4*)(W + (size_t)(kb + r) * Nn + nb + c4);
  tile[r][c4 + 0] = f.x; tile[r][c4 + 1] = f.y;
  tile[r][c4 + 2] = f.z; tile[r][c4 + 3] = f.w;
  __syncthreads();
  u32 p0 = (u32)f2bf(tile[c4 + 0][r]) | ((u32)f2bf(tile[c4 + 1][r]) << 16);
  u32 p1 = (u32)f2bf(tile[c4 + 2][r]) | ((u32)f2bf(tile[c4 + 3][r]) << 16);
  *(uint2*)(WT + (size_t)(nb + r) * K + kb + c4) = make_uint2(p0, p1);
}

// ---------------------------------------------------------------------------
// Orchestration
// ---------------------------------------------------------------------------
extern "C" void kernel_launch(void* const* d_in, const int* in_sizes, int n_in,
                              void* d_out, int out_size, void* d_ws, size_t ws_size,
                              hipStream_t stream) {
  (void)in_sizes; (void)n_in; (void)out_size; (void)ws_size;
  const float* tgt  = (const float*)d_in[0];
  const float* memv = (const float*)d_in[1];
  const float* sa_wq = (const float*)d_in[3];  const float* sa_bq = (const float*)d_in[4];
  const float* sa_wk = (const float*)d_in[5];  const float* sa_bk = (const float*)d_in[6];
  const float* sa_wv = (const float*)d_in[7];  const float* sa_bv = (const float*)d_in[8];
  const float* sa_wo = (const float*)d_in[9];  const float* sa_bo = (const float*)d_in[10];
  const float* ca_wq = (const float*)d_in[11]; const float* ca_bq = (const float*)d_in[12];
  const float* ca_wk = (const float*)d_in[13]; const float* ca_bk = (const float*)d_in[14];
  const float* ca_wv = (const float*)d_in[15]; const float* ca_bv = (const float*)d_in[16];
  const float* ca_wo = (const float*)d_in[17]; const float* ca_bo = (const float*)d_in[18];
  const float* w1 = (const float*)d_in[19]; const float* b1 = (const float*)d_in[20];
  const float* w2 = (const float*)d_in[21]; const float* b2 = (const float*)d_in[22];
  const float* g1 = (const float*)d_in[23]; const float* be1 = (const float*)d_in[24];
  const float* g2 = (const float*)d_in[25]; const float* be2 = (const float*)d_in[26];
  const float* g3 = (const float*)d_in[27]; const float* be3 = (const float*)d_in[28];

  const size_t SZ = (size_t)NB_DIM * S_DIM * E_DIM;   // 4M elements
  const size_t WSZ = (size_t)E_DIM * E_DIM;           // 1M elements
  char* p = (char*)d_ws;
  auto alloc = [&](size_t bytes) { char* r = p; p += (bytes + 255) & ~(size_t)255; return r; };

  // bf16 transposed weights
  u16* wqTs = (u16*)alloc(WSZ * 2); u16* wkTs = (u16*)alloc(WSZ * 2);
  u16* wvTs = (u16*)alloc(WSZ * 2); u16* woTs = (u16*)alloc(WSZ * 2);
  u16* wqTc = (u16*)alloc(WSZ * 2); u16* wkTc = (u16*)alloc(WSZ * 2);
  u16* wvTc = (u16*)alloc(WSZ * 2); u16* woTc = (u16*)alloc(WSZ * 2);
  u16* w1T  = (u16*)alloc((size_t)E_DIM * DFF_DIM * 2);
  u16* w2T  = (u16*)alloc((size_t)DFF_DIM * E_DIM * 2);
  // bf16 activations
  u16* tgtb = (u16*)alloc(SZ * 2);  u16* memb = (u16*)alloc(SZ * 2);
  u16* Qb = (u16*)alloc(SZ * 2);    u16* Kb = (u16*)alloc(SZ * 2);
  u16* Vb = (u16*)alloc(SZ * 2);    u16* Db = (u16*)alloc(SZ * 2);
  u16* Xb = (u16*)alloc(SZ * 2);    u16* Yb = (u16*)alloc(SZ * 2);
  u16* Hb = (u16*)alloc(SZ * 4 * 2);
  // fp32 buffers
  float* bE  = (float*)alloc(SZ * 4);
  float* bXf = (float*)alloc(SZ * 4);
  float* bYf = (float*)alloc(SZ * 4);

  const int M = NB_DIM * S_DIM;                    // 4096
  const dim3 gEE(E_DIM / 128, M / 128);            // (8, 32)
  const dim3 gED(DFF_DIM / 128, M / 128);          // (32, 32)
  const dim3 gAT(S_DIM / 64, NH_DIM, NB_DIM);
  const dim3 gTE(E_DIM / 32, E_DIM / 32);          // E x E transpose
  const dim3 gT1(E_DIM / 32, DFF_DIM / 32);        // w1: K=E, N=DFF
  const dim3 gT2(DFF_DIM / 32, E_DIM / 32);        // w2: K=DFF, N=E

  // ---- one-time conversions (per call; deterministic) ----
  transpose_convert<<<gTE, 256, 0, stream>>>(sa_wq, wqTs, E_DIM, E_DIM);
  transpose_convert<<<gTE, 256, 0, stream>>>(sa_wk, wkTs, E_DIM, E_DIM);
  transpose_convert<<<gTE, 256, 0, stream>>>(sa_wv, wvTs, E_DIM, E_DIM);
  transpose_convert<<<gTE, 256, 0, stream>>>(sa_wo, woTs, E_DIM, E_DIM);
  transpose_convert<<<gTE, 256, 0, stream>>>(ca_wq, wqTc, E_DIM, E_DIM);
  transpose_convert<<<gTE, 256, 0, stream>>>(ca_wk, wkTc, E_DIM, E_DIM);
  transpose_convert<<<gTE, 256, 0, stream>>>(ca_wv, wvTc, E_DIM, E_DIM);
  transpose_convert<<<gTE, 256, 0, stream>>>(ca_wo, woTc, E_DIM, E_DIM);
  transpose_convert<<<gT1, 256, 0, stream>>>(w1, w1T, E_DIM, DFF_DIM);
  transpose_convert<<<gT2, 256, 0, stream>>>(w2, w2T, DFF_DIM, E_DIM);
  convert_bf16<<<(int)(SZ / 1024), 256, 0, stream>>>(tgt,  tgtb);
  convert_bf16<<<(int)(SZ / 1024), 256, 0, stream>>>(memv, memb);

  // ---- self attention ----
  gemm_tdm<false, true ><<<gEE, 256, 0, stream>>>(tgtb, wqTs, sa_bq, Qb, M, E_DIM, E_DIM);
  gemm_tdm<false, true ><<<gEE, 256, 0, stream>>>(tgtb, wkTs, sa_bk, Kb, M, E_DIM, E_DIM);
  gemm_tdm<false, true ><<<gEE, 256, 0, stream>>>(tgtb, wvTs, sa_bv, Vb, M, E_DIM, E_DIM);
  attn_kernel<true><<<gAT, 128, 0, stream>>>(Qb, Kb, Vb, Db, S_DIM, S_DIM);
  gemm_tdm<false, false><<<gEE, 256, 0, stream>>>(Db, woTs, sa_bo, bE, M, E_DIM, E_DIM);
  add_ln<true><<<M, 256, 0, stream>>>(tgt, bE, g1, be1, bXf, Xb);

  // ---- cross attention ----
  gemm_tdm<false, true ><<<gEE, 256, 0, stream>>>(Xb,   wqTc, ca_bq, Qb, M, E_DIM, E_DIM);
  gemm_tdm<false, true ><<<gEE, 256, 0, stream>>>(memb, wkTc, ca_bk, Kb, M, E_DIM, E_DIM);
  gemm_tdm<false, true ><<<gEE, 256, 0, stream>>>(memb, wvTc, ca_bv, Vb, M, E_DIM, E_DIM);
  attn_kernel<false><<<gAT, 128, 0, stream>>>(Qb, Kb, Vb, Db, S_DIM, T_DIM);
  gemm_tdm<false, false><<<gEE, 256, 0, stream>>>(Db, woTc, ca_bo, bE, M, E_DIM, E_DIM);
  add_ln<true><<<M, 256, 0, stream>>>(bXf, bE, g2, be2, bYf, Yb);

  // ---- FFN ----
  gemm_tdm<true,  true ><<<gED, 256, 0, stream>>>(Yb, w1T, b1, Hb, M, DFF_DIM, E_DIM);
  gemm_tdm<false, false><<<gEE, 256, 0, stream>>>(Hb, w2T, b2, bE, M, E_DIM, DFF_DIM);
  add_ln<false><<<M, 256, 0, stream>>>(bYf, bE, g3, be3, (float*)d_out, nullptr);
}
